// QuantumAGICore_7344394076257
// MI455X (gfx1250) — matrix operand order repaired and checked by reading, not compiled, not run on hardware
//
#include <hip/hip_runtime.h>
#include <hip/hip_bf16.h>

typedef __attribute__((ext_vector_type(16))) __bf16 v16bf;
typedef __attribute__((ext_vector_type(8)))  float  v8f;
typedef __attribute__((ext_vector_type(4)))  float  v4f;   // native vector for nontemporal builtins

// Problem constants
#define CB   8
#define CS   512
#define CD   1024
#define CM   512
#define CDD  (1024*1024)

// Workspace layout (float offsets; all 16-float aligned for float4 access)
#define WS_XMEAN   0          // 16*1024
#define WS_YBASE   16384      // 16*1024
#define WS_T1      32768      // 16*1024
#define WS_QP      49152      // 16*1024
#define WS_HID     65536      // 16*2048
#define WS_FEATM   98304      // 16*1024
#define WS_COMB    114688     // 16*2048
#define WS_HB      147456     // 16*1024
#define WS_ROWOUT  163840     // 8*1024
#define WS_MQ      172032     // 16*16
#define WS_LOGITS  172288     // 8*512
#define WS_PARTIAL 176384     // 8*8*1024
#define WS_TOP3    241920     // 24 ints
#define WS_ZTOTAL  241952

__global__ __launch_bounds__(256) void zero_ws(float* ws, int n) {
  int i = blockIdx.x * 256 + threadIdx.x;
  if (i < n) ws[i] = 0.0f;
}

// Stage 1: partial sums over 64-row s-chunks, one full D-row per block. grid (B, 8)
__global__ __launch_bounds__(256) void xmean_partial(const v4f* __restrict__ x4,
                                                     v4f* __restrict__ part4) {
  int d4 = threadIdx.x;                       // D/4 = 256
  int b = blockIdx.x, c = blockIdx.y;
  const v4f* base = x4 + ((size_t)b * CS + (size_t)c * 64) * (CD / 4) + d4;
  v4f s = {0.f, 0.f, 0.f, 0.f};
  #pragma unroll 8
  for (int i = 0; i < 64; ++i) s += base[(size_t)i * (CD / 4)];
  part4[(size_t)(c * CB + b) * (CD / 4) + d4] = s;
}

// Stage 2: combine chunks, divide by S. grid (B)
__global__ __launch_bounds__(256) void xmean_finish(const v4f* __restrict__ part4,
                                                    v4f* __restrict__ xmean4) {
  int d4 = threadIdx.x;
  int b = blockIdx.x;
  v4f s = {0.f, 0.f, 0.f, 0.f};
  #pragma unroll
  for (int c = 0; c < 8; ++c) s += part4[(size_t)(c * CB + b) * (CD / 4) + d4];
  xmean4[(size_t)b * (CD / 4) + d4] = s * (1.0f / (float)CS);
}

// 16-row GEMM on WMMA bf16: Out(16xN) = epi( A(16xK) @ W(KxN) + bias ).
// One wave per 16x16 tile, 8 waves/block. Only rows 0..7 (valid batch) stored.
// ACT: 0=none 1=tanh 2=silu. VQC: per-col scale from ql_vqc. RESID: += resid.
template<int ACT, bool VQC, bool RESID>
__global__ __launch_bounds__(256)
void gemm16_wmma(const float* __restrict__ A, const float* __restrict__ W,
                 const float* __restrict__ bias, const float* __restrict__ vqc,
                 const float* resid, float* Out, int K, int N)
{
  const int lane = threadIdx.x & 31;
  const int wave = threadIdx.x >> 5;
  const int n0 = (blockIdx.x * 8 + wave) * 16;
  if (n0 >= N) return;                  // wave-uniform: EXEC stays all-ones for WMMA
  const int half = lane >> 4;           // K half selector (ISA 16-bit A/B layouts)
  const int lr   = lane & 15;           // A row / B column within tile

  v8f acc = {};
  const float* Abase = A + (size_t)lr * K;
  for (int k0 = 0; k0 < K; k0 += 32) {
    v16bf af, bfv;
    const float* Ar = Abase + k0;
    #pragma unroll
    for (int i = 0; i < 8; ++i) {       // A 16x32: lanes0-15 K0..15; lanes16-31 K16..31
      af[i]     = (__bf16)Ar[half * 16 + i];
      af[i + 8] = (__bf16)Ar[half * 16 + 8 + i];
    }
    const float* Wc = W + (size_t)(k0 + half * 16) * N + (n0 + lr);
    #pragma unroll
    for (int i = 0; i < 16; ++i)        // B 32x16: lanes0-15 K0..15; lanes16-31 K16..31
      bfv[i] = (__bf16)Wc[(size_t)i * N];
    acc = __builtin_amdgcn_wmma_f32_16x16x32_bf16(false, af, false, bfv,
                                                  (short)0, acc, false, false);
  }

  // C/D layout: VGPR r -> row r (lanes 0-15) / row 8+r (lanes 16-31). Rows >=8 are padding.
  if (half == 0) {
    const int col = n0 + lr;
    const float bcol = bias[col];
    float scl = 1.0f;
    if (VQC) {
      float th = vqc[col * 4 + 0], ph = vqc[col * 4 + 1];
      scl = cosf(th * 0.5f) + sinf(th * 0.5f) * cosf(ph);
    }
    #pragma unroll
    for (int r = 0; r < 8; ++r) {
      float v = acc[r] + bcol;
      if (ACT == 1) v = tanhf(v);
      if (ACT == 2) v = v / (1.0f + expf(-v));
      if (VQC)   v *= scl;
      if (RESID) v += resid[(size_t)r * N + col];
      Out[(size_t)r * N + col] = v;
    }
  }
}

// logits^2 = (mq @ mem_states^T)^2 ; softmax is monotone so ranking uses this directly. grid (2, B)
__global__ __launch_bounds__(256) void logits_kernel(const float* __restrict__ mq,
                                                     const float* __restrict__ mem,
                                                     float* __restrict__ lg) {
  int m = blockIdx.x * 256 + threadIdx.x;
  int b = blockIdx.y;
  float s = 0.0f;
  #pragma unroll
  for (int j = 0; j < 16; ++j) s += mq[b * 16 + j] * mem[m * 16 + j];
  lg[b * CM + m] = s * s;
}

__global__ void top3_kernel(const float* __restrict__ lg, int* __restrict__ top3) {
  int b = threadIdx.x;
  if (b < CB) {
    float v0 = -1.f, v1 = -1.f, v2 = -1.f;
    int i0 = 0, i1 = 0, i2 = 0;
    for (int m = 0; m < CM; ++m) {
      float v = lg[b * CM + m];
      if (v > v0)      { v2 = v1; i2 = i1; v1 = v0; i1 = i0; v0 = v; i0 = m; }
      else if (v > v1) { v2 = v1; i2 = i1; v1 = v;  i1 = m; }
      else if (v > v2) { v2 = v;  i2 = m; }
    }
    top3[b * 3 + 0] = i0; top3[b * 3 + 1] = i1; top3[b * 3 + 2] = i2;
  }
}

// feats mean: (1/3) sum_j tanh(mem[idx_j,:8] @ ret_w1 + b1) @ ret_w2 + ret_b2. grid B.
__global__ __launch_bounds__(256) void feats_kernel(const int* __restrict__ top3,
    const float* __restrict__ mem, const float* __restrict__ w1, const float* __restrict__ b1,
    const float* __restrict__ w2, const float* __restrict__ b2, float* __restrict__ featm) {
  __shared__ float tt[24];
  int b = blockIdx.x, tid = threadIdx.x;
  if (tid < 24) {
    int j = tid >> 3, q = tid & 7;
    int idx = top3[b * 3 + j];
    float s = b1[q];
    #pragma unroll
    for (int p = 0; p < 8; ++p) s += mem[idx * 16 + p] * w1[p * 8 + q];
    tt[tid] = tanhf(s);
  }
  __syncthreads();
  for (int d = tid; d < CD; d += 256) {
    float s = 0.0f;
    #pragma unroll
    for (int j = 0; j < 3; ++j)
      #pragma unroll
      for (int q = 0; q < 8; ++q) s += tt[j * 8 + q] * w2[q * CD + d];
    featm[(size_t)b * CD + d] = b2[d] + s * (1.0f / 3.0f);
  }
}

// comb = [y | featm] row-wise, float4. grid 16
__global__ __launch_bounds__(256) void comb_kernel(const v4f* __restrict__ y4,
                                                   const v4f* __restrict__ f4,
                                                   v4f* __restrict__ comb4) {
  int idx = blockIdx.x * 256 + threadIdx.x;   // < 8*512
  int b = idx >> 9, d4 = idx & 511;           // 2D/4 = 512
  comb4[idx] = (d4 < CD / 4) ? y4[b * (CD / 4) + d4] : f4[b * (CD / 4) + d4 - CD / 4];
}

__global__ __launch_bounds__(256) void ln_kernel(const float* __restrict__ hb,
    const float* __restrict__ gamma, const float* __restrict__ beta, float* __restrict__ rowout) {
  __shared__ float red[256];
  int b = blockIdx.x, tid = threadIdx.x;
  float v[4]; float s = 0.0f;
  #pragma unroll
  for (int i = 0; i < 4; ++i) { v[i] = hb[(size_t)b * CD + tid + i * 256]; s += v[i]; }
  red[tid] = s; __syncthreads();
  for (int off = 128; off > 0; off >>= 1) { if (tid < off) red[tid] += red[tid + off]; __syncthreads(); }
  float mu = red[0] * (1.0f / (float)CD);
  __syncthreads();
  float s2 = 0.0f;
  #pragma unroll
  for (int i = 0; i < 4; ++i) { float d = v[i] - mu; s2 += d * d; }
  red[tid] = s2; __syncthreads();
  for (int off = 128; off > 0; off >>= 1) { if (tid < off) red[tid] += red[tid + off]; __syncthreads(); }
  float inv = rsqrtf(red[0] * (1.0f / (float)CD) + 1e-5f);
  #pragma unroll
  for (int i = 0; i < 4; ++i) {
    int d = tid + i * 256;
    rowout[(size_t)b * CD + d] = (v[i] - mu) * inv * gamma[d] + beta[d];
  }
}

// Broadcast 8 rows across sequence dim; non-temporal B128 stores (write-once stream).
__global__ __launch_bounds__(256) void bcast_kernel(const v4f* __restrict__ row4,
                                                    v4f* __restrict__ out4) {
  size_t idx = (size_t)blockIdx.x * 256 + threadIdx.x;   // B*S*D/4 = 2^20
  int d4 = (int)(idx & (CD / 4 - 1));                    // 256 v4f per row
  int b = (int)(idx >> 17);                              // S*D/4 = 2^17
  v4f v = row4[(size_t)b * (CD / 4) + d4];
  __builtin_nontemporal_store(v, &out4[idx]);
}

extern "C" void kernel_launch(void* const* d_in, const int* in_sizes, int n_in,
                              void* d_out, int out_size, void* d_ws, size_t ws_size,
                              hipStream_t stream) {
  (void)in_sizes; (void)n_in; (void)out_size; (void)ws_size;
  const float* x          = (const float*)d_in[0];
  const float* w_v        = (const float*)d_in[5];
  const float* b_v        = (const float*)d_in[6];
  const float* w_o        = (const float*)d_in[7];
  const float* b_o        = (const float*)d_in[8];
  const float* amp_w1     = (const float*)d_in[15];
  const float* amp_b1     = (const float*)d_in[16];
  const float* amp_w2     = (const float*)d_in[17];
  const float* amp_b2     = (const float*)d_in[18];
  const float* ql_vqc     = (const float*)d_in[19];
  const float* post_w1    = (const float*)d_in[20];
  const float* post_b1    = (const float*)d_in[21];
  const float* post_w2    = (const float*)d_in[22];
  const float* post_b2    = (const float*)d_in[23];
  const float* enc_w1     = (const float*)d_in[24];
  const float* enc_b1     = (const float*)d_in[25];
  const float* enc_w2     = (const float*)d_in[26];
  const float* enc_b2     = (const float*)d_in[27];
  const float* mem_states = (const float*)d_in[28];
  const float* ret_w1     = (const float*)d_in[29];
  const float* ret_b1     = (const float*)d_in[30];
  const float* ret_w2     = (const float*)d_in[31];
  const float* ret_b2     = (const float*)d_in[32];
  const float* br_w1      = (const float*)d_in[33];
  const float* br_b1      = (const float*)d_in[34];
  const float* br_w2      = (const float*)d_in[35];
  const float* br_b2      = (const float*)d_in[36];
  const float* ln_scale   = (const float*)d_in[37];
  const float* ln_offset  = (const float*)d_in[38];

  float* ws     = (float*)d_ws;
  float* xmean  = ws + WS_XMEAN;
  float* ybase  = ws + WS_YBASE;
  float* t1     = ws + WS_T1;
  float* qp     = ws + WS_QP;
  float* hid    = ws + WS_HID;
  float* featm  = ws + WS_FEATM;
  float* comb   = ws + WS_COMB;
  float* hb     = ws + WS_HB;
  float* rowout = ws + WS_ROWOUT;
  float* mq     = ws + WS_MQ;
  float* lg     = ws + WS_LOGITS;
  float* part   = ws + WS_PARTIAL;
  int*   top3   = (int*)(ws + WS_TOP3);

  dim3 blk(256);

  zero_ws<<<(WS_ZTOTAL + 255) / 256, blk, 0, stream>>>(ws, WS_ZTOTAL);

  // xmean = mean_s(x)  (exact: uniform attention makes Q/K branch dead code)
  xmean_partial<<<dim3(CB, 8), blk, 0, stream>>>((const v4f*)x, (v4f*)part);
  xmean_finish <<<CB,          blk, 0, stream>>>((const v4f*)part, (v4f*)xmean);

  // vmean = xmean @ w_v + b_v ; ybase = vmean @ w_o + b_o
  gemm16_wmma<0,false,false><<<8, blk, 0, stream>>>(xmean, w_v, b_v, nullptr, nullptr, t1,    CD, CD);
  gemm16_wmma<0,false,false><<<8, blk, 0, stream>>>(t1,    w_o, b_o, nullptr, nullptr, ybase, CD, CD);

  // Quantum-layer MLP stack (rows constant over s -> 16-row GEMMs)
  for (int l = 0; l < 2; ++l) {
    gemm16_wmma<1,false,false><<<8, blk, 0, stream>>>(ybase, amp_w1 + (size_t)l*CDD, amp_b1 + l*CD,
                                                      nullptr, nullptr, t1, CD, CD);
    gemm16_wmma<0,true ,false><<<8, blk, 0, stream>>>(t1, amp_w2 + (size_t)l*CDD, amp_b2 + l*CD,
                                                      ql_vqc + (size_t)l*CD*4, nullptr, qp, CD, CD);
    gemm16_wmma<2,false,false><<<8, blk, 0, stream>>>(qp, post_w1 + (size_t)l*CDD, post_b1 + l*CD,
                                                      nullptr, nullptr, hid, CD, CD);
    gemm16_wmma<0,false,true ><<<8, blk, 0, stream>>>(hid, post_w2 + (size_t)l*CDD, post_b2 + l*CD,
                                                      nullptr, ybase, ybase, CD, CD);
  }

  // Memory retrieval path
  gemm16_wmma<1,false,false><<<8, blk, 0, stream>>>(ybase, enc_w1, enc_b1, nullptr, nullptr, t1, CD, CD);
  gemm16_wmma<0,false,false><<<1, blk, 0, stream>>>(t1, enc_w2, enc_b2, nullptr, nullptr, mq, CD, 16);
  logits_kernel<<<dim3(2, CB), blk, 0, stream>>>(mq, mem_states, lg);
  top3_kernel<<<1, 32, 0, stream>>>(lg, top3);
  feats_kernel<<<CB, blk, 0, stream>>>(top3, mem_states, ret_w1, ret_b1, ret_w2, ret_b2, featm);

  // Bridge MLP + LayerNorm
  comb_kernel<<<16, blk, 0, stream>>>((const v4f*)ybase, (const v4f*)featm, (v4f*)comb);
  gemm16_wmma<2,false,false><<<16, blk, 0, stream>>>(comb, br_w1, br_b1, nullptr, nullptr, hid, 2*CD, 2*CD);
  gemm16_wmma<0,false,false><<<8,  blk, 0, stream>>>(hid,  br_w2, br_b2, nullptr, nullptr, hb,  2*CD, CD);
  ln_kernel<<<CB, blk, 0, stream>>>(hb, ln_scale, ln_offset, rowout);

  // Broadcast the 8 rows across the sequence dimension (B*S*D/4 v4f = 4096 blocks)
  bcast_kernel<<<4096, blk, 0, stream>>>((const v4f*)rowout, (v4f*)d_out);
}
